// TGPKernel_21036749815950
// MI455X (gfx1250) — matrix operand order repaired and checked by reading, not compiled, not run on hardware
//
#include <hip/hip_runtime.h>
#include <hip/hip_bf16.h>

typedef __attribute__((ext_vector_type(16))) __bf16 v16bf;
typedef __attribute__((ext_vector_type(8)))  float  v8f;
typedef __attribute__((ext_vector_type(8)))  int    v8i_t;

#define N1R 8192
#define N2R 8192
#define DD  128
#define KT  4        // 128 / 32 k-tiles

// Fragment storage: tile-major, 32 lanes x 16 bf16 per tile (matches WMMA VGPR image)
__device__ __forceinline__ size_t frag_off(int tile, int lane) {
    return ((size_t)tile * 32 + lane) * 16;
}

// mode 0: A-operand swizzle (16x32 M-major, K interleaved {0..7,16..23}/{8..15,24..31})
// mode 1: B-operand swizzle (32x16 col-per-lane, K interleaved {0..7,16..23}/{8..15,24..31})
__global__ void prep_kernel(const float* __restrict__ X, const float* __restrict__ log_l,
                            __bf16* __restrict__ hi, __bf16* __restrict__ lo,
                            int mode)
{
    int tid  = blockIdx.x * blockDim.x + threadIdx.x;
    int lane = tid & 31;
    int kt   = (tid >> 5) & (KT - 1);
    int mt   = tid >> 7;                 // tid / (32*KT)
    int row  = mt * 16 + (lane & 15);    // row of X (A: M row, B: X2 row = N column)

    const float* xr = X + (size_t)row * DD;
    v16bf hv, lv;
    if (mode == 0) {
        int kbase = kt * 32 + ((lane >= 16) ? 8 : 0);
        #pragma unroll
        for (int j = 0; j < 16; j++) {
            int k = kbase + j + ((j >= 8) ? 8 : 0);
            float a = xr[k] * __expf(-log_l[k]);
            __bf16 ah = (__bf16)a;
            hv[j] = ah;
            lv[j] = (__bf16)(a - (float)ah);
        }
    } else {
        int kbase = kt * 32 + ((lane >= 16) ? 8 : 0);
        #pragma unroll
        for (int j = 0; j < 16; j++) {
            int k = kbase + j + ((j >= 8) ? 8 : 0);
            float a = xr[k] * __expf(-log_l[k]);
            __bf16 ah = (__bf16)a;
            hv[j] = ah;
            lv[j] = (__bf16)(a - (float)ah);
        }
    }
    size_t off = frag_off(mt * KT + kt, lane);
    *(v16bf*)(hi + off) = hv;
    *(v16bf*)(lo + off) = lv;
}

__global__ void rownorm_kernel(const float* __restrict__ X, const float* __restrict__ log_l,
                               float* __restrict__ norms, int nrows)
{
    int r = blockIdx.x * blockDim.x + threadIdx.x;
    if (r >= nrows) return;
    const float* xr = X + (size_t)r * DD;
    float s = 0.0f;
    #pragma unroll 8
    for (int k = 0; k < DD; k++) {
        float a = xr[k] * __expf(-log_l[k]);
        s += a * a;
    }
    norms[r] = s;
}

// Block: 256 threads = 8 waves arranged 2(M) x 4(N). Block tile 128x128.
// Wave tile: MT=4 m-tiles x NT=2 n-tiles. B fragments (hi+lo, both n-tiles,
// all 4 k-tiles) stay resident in VGPRs; each A fragment load feeds 6 WMMAs.
__global__ void __launch_bounds__(256)
gemm_rbf_kernel(const __bf16* __restrict__ Ahi, const __bf16* __restrict__ Alo,
                const __bf16* __restrict__ Bhi, const __bf16* __restrict__ Blo,
                const float* __restrict__ n1v, const float* __restrict__ n2v,
                const int* __restrict__ task1, const int* __restrict__ task2,
                const float* __restrict__ log_theta, const float* __restrict__ bparam,
                float* __restrict__ out)
{
    const int lane = threadIdx.x & 31;
    const int w    = threadIdx.x >> 5;
    const int wn   = w & 3;                       // 4 waves along N
    const int wm   = w >> 2;                      // 2 waves along M
    const int mtb  = blockIdx.x * 8 + wm * 4;     // first M tile of this wave
    const int ntb  = blockIdx.y * 8 + wn * 2;     // first N tile of this wave

    // Resident B fragments: 2 n-tiles x 4 k-tiles x (hi,lo)
    v16bf bh[2][KT], bl[2][KT];
    #pragma unroll
    for (int j = 0; j < 2; j++) {
        #pragma unroll
        for (int kt = 0; kt < KT; kt++) {
            size_t off = frag_off((ntb + j) * KT + kt, lane);
            bh[j][kt] = *(const v16bf*)(Bhi + off);
            bl[j][kt] = *(const v16bf*)(Blo + off);
        }
    }

    // theta*exp(-0.5*max(sq,0)) == exp2( min(CEXP*sq + lt2, lt2) )
    const float LOG2E = 1.4426950408889634f;
    const float CEXP  = -0.72134752044448f;       // -0.5 * log2(e)
    const float lt2   = log_theta[0] * LOG2E;     // log2(theta)
    float lam = 2.0f / (1.0f + __expf(bparam[0])) - 1.0f;
    lam = fminf(fmaxf(lam, 0.0f), 1.0f);

    const int hihalf = (lane >= 16) ? 8 : 0;
    float  nrm2[2];
    int    t2[2];
    float* rowp[2];   // per n-tile output pointer for current m-tile row block
    #pragma unroll
    for (int j = 0; j < 2; j++) {
        int ncol = (ntb + j) * 16 + (lane & 15);
        nrm2[j]  = n2v[ncol];
        t2[j]    = task2[ncol];
        rowp[j]  = out + (size_t)(mtb * 16 + hihalf) * N2R + ncol;
    }

    #pragma unroll
    for (int mt = 0; mt < 4; mt++) {
        v8f acc[2] = {{}, {}};
        #pragma unroll
        for (int kt = 0; kt < KT; kt++) {
            size_t off = frag_off((mtb + mt) * KT + kt, lane);
            v16bf ah = *(const v16bf*)(Ahi + off);
            v16bf al = *(const v16bf*)(Alo + off);
            #pragma unroll
            for (int j = 0; j < 2; j++) {
                acc[j] = __builtin_amdgcn_wmma_f32_16x16x32_bf16(false, ah, false, bh[j][kt],
                                                                 (short)0, acc[j], false, false);
                acc[j] = __builtin_amdgcn_wmma_f32_16x16x32_bf16(false, ah, false, bl[j][kt],
                                                                 (short)0, acc[j], false, false);
                acc[j] = __builtin_amdgcn_wmma_f32_16x16x32_bf16(false, al, false, bh[j][kt],
                                                                 (short)0, acc[j], false, false);
            }
        }
        // Epilogue: C layout -> VGPR v: M = v + 8*(lane>=16), N = lane%16.
        // Rows v=0..7 are contiguous from mbase (32B aligned): one v8f + one v8i load.
        const int mbase = (mtb + mt) * 16 + hihalf;
        v8f   n1frag = *(const v8f*)(n1v + mbase);
        v8i_t t1frag = *(const v8i_t*)(task1 + mbase);
        #pragma unroll
        for (int j = 0; j < 2; j++) {
            #pragma unroll
            for (int v = 0; v < 8; v++) {
                // e = CEXP*(n1+n2-2*dot) + log2(theta), clamped at sq>=0 (e<=lt2)
                float e = fmaf(LOG2E, acc[j][v],
                               fmaf(CEXP, n1frag[v] + nrm2[j], lt2));
                e = fminf(e, lt2);
                float val = exp2f(e);
                if (t1frag[v] != t2[j]) val *= lam;
                // v*N2R is a compile-time byte offset (v*32768 < 2^23): folds
                // into the store's immediate offset off the loop-carried pointer.
                __builtin_nontemporal_store(val, rowp[j] + (size_t)v * N2R);
            }
            rowp[j] += (size_t)16 * N2R;   // advance to next m-tile row block
        }
    }
}

extern "C" void kernel_launch(void* const* d_in, const int* in_sizes, int n_in,
                              void* d_out, int out_size, void* d_ws, size_t ws_size,
                              hipStream_t stream) {
    const float* X1        = (const float*)d_in[0];
    const float* X2        = (const float*)d_in[1];
    const float* log_l     = (const float*)d_in[2];
    const float* log_theta = (const float*)d_in[3];
    const float* bparam    = (const float*)d_in[4];
    const int*   task1     = (const int*)d_in[5];
    const int*   task2     = (const int*)d_in[6];
    float*       out       = (float*)d_out;

    // workspace layout
    __bf16* Ahi = (__bf16*)d_ws;
    __bf16* Alo = Ahi + (size_t)N1R * DD;
    __bf16* Bhi = Alo + (size_t)N1R * DD;
    __bf16* Blo = Bhi + (size_t)N2R * DD;
    float*  n1v = (float*)(Blo + (size_t)N2R * DD);
    float*  n2v = n1v + N1R;

    const int fragThreadsA = (N1R / 16) * KT * 32;   // 65536
    const int fragThreadsB = (N2R / 16) * KT * 32;   // 65536
    prep_kernel<<<fragThreadsA / 256, 256, 0, stream>>>(X1, log_l, Ahi, Alo, 0);
    prep_kernel<<<fragThreadsB / 256, 256, 0, stream>>>(X2, log_l, Bhi, Blo, 1);
    rownorm_kernel<<<N1R / 256, 256, 0, stream>>>(X1, log_l, n1v, N1R);
    rownorm_kernel<<<N2R / 256, 256, 0, stream>>>(X2, log_l, n2v, N2R);

    dim3 grid(N1R / 128, N2R / 128);   // 64 x 64 blocks, 128x128 tile each
    gemm_rbf_kernel<<<grid, 256, 0, stream>>>(Ahi, Alo, Bhi, Blo,
                                              n1v, n2v, task1, task2,
                                              log_theta, bparam, out);
}
